// MambaLayer_2705829396804
// MI455X (gfx1250) — compile-verified, matrix operand order checked by reading, not run on hardware
//
#include <hip/hip_runtime.h>
#include <hip/hip_bf16.h>

// ---------------------------------------------------------------------------
// Mamba-3D (Vim-style, 3 scan directions) for MI455X / gfx1250.
// GEMMs: v_wmma_f32_16x16x32_bf16, weight tile staged in LDS per workgroup
// (8 waves share it). Inner loop: issue all A/B fragment loads as a clause,
// then NT back-to-back WMMAs (distinct regs -> single wait per k-step).
// Scan: 1536 wave32s (3 branches x B x Di/2), 2 channels x 16 states each.
// ---------------------------------------------------------------------------

typedef __attribute__((ext_vector_type(16))) __bf16 v16bf;
typedef __attribute__((ext_vector_type(8)))  float  v8f;

static constexpr int BB    = 2;
static constexpr int CC    = 256;
static constexpr int LTOT  = 4096;        // 16*16*16
static constexpr int DI    = 512;         // d_inner
static constexpr int E2    = 1024;        // 2*d_inner
static constexpr int NST   = 16;          // d_state
static constexpr int DTR   = 16;          // dt_rank
static constexpr int XDL   = DTR + 2*NST; // 48
static constexpr int NSL   = 16;          // n slices
static constexpr int LC    = LTOT / NSL;  // 256
static constexpr int MROWS = BB * LTOT;   // 8192

struct U4 { unsigned int x, y, z, w; };
union V16 { v16bf v; U4 q[2]; };

__device__ __forceinline__ float  bf2f(__bf16 v) { return (float)v; }
__device__ __forceinline__ __bf16 f2bf(float v)  { return (__bf16)v; }

// branch position mapping: branch-seq position t -> index into canonical xz rows
__device__ __forceinline__ int map_idx(int br, int t) {
  if (br == 0) return t;                       // forward
  if (br == 1) return LTOT - 1 - t;            // backward (flip)
  return (t & (NSL - 1)) * LC + (t >> 4);      // slice interleave: p=i*16+s <- s*256+i
}

// ---------------------------------------------------------------------------
__global__ void zero_f32(float* p, int n) {
  int i = blockIdx.x * blockDim.x + threadIdx.x;
  if (i < n) p[i] = 0.f;
}

__global__ void cvt_f32_bf16(const float* __restrict__ s, __bf16* __restrict__ d, int n) {
  int i = blockIdx.x * blockDim.x + threadIdx.x;
  if (i < n) d[i] = f2bf(s[i]);
}

// LayerNorm over C + transpose (B,C,L) -> (B*L, C) bf16
__global__ void ln_kernel(const float* __restrict__ x, const float* __restrict__ w,
                          const float* __restrict__ b, __bf16* __restrict__ xn) {
  int row = blockIdx.x;                 // b*L + l
  int bb = row / LTOT, l = row % LTOT;
  int c = threadIdx.x;                  // 256 threads
  float v = x[((size_t)bb * CC + c) * LTOT + l];
  __shared__ float s1[CC], s2[CC];
  s1[c] = v; s2[c] = v * v;
  __syncthreads();
  for (int s = CC / 2; s > 0; s >>= 1) {
    if (c < s) { s1[c] += s1[c + s]; s2[c] += s2[c + s]; }
    __syncthreads();
  }
  float mu  = s1[0] * (1.f / CC);
  float var = s2[0] * (1.f / CC) - mu * mu;
  float r   = rsqrtf(var + 1e-5f);
  xn[(size_t)row * CC + c] = f2bf((v - mu) * r * w[c] + b[c]);
}

// ---------------------------------------------------------------------------
// WMMA GEMM: Out[m][n] = sum_k A[m][k] * W[n][k]   (A,W bf16 row-major, ldW=K)
// Block = 8 waves = 128(M) x NT*16(N) macro-tile; W tile staged in LDS once.
// MODE 0: store bf16 (ldO)     MODE 1: store f32 (ldO)
// MODE 2: out_proj epilogue -> d_out[b][c][l] = acc + resid[b][c][l]
// Per-lane fragment layout per ISA 7.12.2 (16-bit A 16x32); B symmetric.
template <int MODE, int K, int NT>
__global__ void __launch_bounds__(256)
gemm_wmma(const __bf16* __restrict__ A, const __bf16* __restrict__ W,
          void* __restrict__ Out, int ldA, int ldO, int nTiles,
          const float* __restrict__ resid, int Cdim, int Ldim) {
  constexpr int NCOLS = NT * 16;
  __shared__ __bf16 Wl[NCOLS * K];

  int mblk = blockIdx.x / nTiles;
  int nt   = blockIdx.x % nTiles;
  int tid  = threadIdx.x;

  // stage contiguous W tile (NCOLS rows x K halfs) into LDS: b128 copies
  {
    const U4* src = (const U4*)(W + (size_t)nt * NCOLS * K);
    U4*       dst = (U4*)Wl;
#pragma unroll
    for (int i = tid; i < NCOLS * K / 8; i += 256) dst[i] = src[i];
  }
  __syncthreads();

  int widx = tid >> 5, lane = tid & 31;
  int g = lane >> 4, r = lane & 15;
  int mt = mblk * 8 + widx;                       // this wave's 16-row m tile

  v8f acc[NT] = {};
  const __bf16* Arow = A + (size_t)(mt * 16 + r) * ldA;
  const __bf16* Wrow0 = Wl + (size_t)r * K;

#pragma unroll 4
  for (int k0 = 0; k0 < K; k0 += 32) {
    if (k0 + 128 < K) __builtin_prefetch(Arow + k0 + 128, 0, 1);  // global_prefetch_b8

    // ---- load phase: A fragment (global b128 x2) + NT B fragments (LDS) ----
    V16 a;
    a.q[0] = *(const U4*)(Arow + k0 + g * 8);        // halfs 0..7 : K = k0 + g*8 ..
    a.q[1] = *(const U4*)(Arow + k0 + 16 + g * 8);   // halfs 8..15: K = k0+16+g*8 ..
    V16 bm[NT];
#pragma unroll
    for (int j = 0; j < NT; ++j) {                   // ds_load_b128 clause
      const __bf16* Wrow = Wrow0 + j * 16 * K;
      bm[j].q[0] = *(const U4*)(Wrow + k0 + g * 8);
      bm[j].q[1] = *(const U4*)(Wrow + k0 + 16 + g * 8);
    }

    // ---- compute phase: NT back-to-back WMMAs (disjoint regs, no hazards) --
#pragma unroll
    for (int j = 0; j < NT; ++j) {
      acc[j] = __builtin_amdgcn_wmma_f32_16x16x32_bf16(
          false, a.v, false, bm[j].v, (short)0, acc[j], false, false);
    }
  }

#pragma unroll
  for (int j = 0; j < NT; ++j) {
    int ncol = nt * NCOLS + j * 16 + r;
#pragma unroll
    for (int rr = 0; rr < 8; ++rr) {
      int m = mt * 16 + g * 8 + rr;                  // C/D layout: VGPR rr, half g
      float v = acc[j][rr];
      if (MODE == 0) {
        ((__bf16*)Out)[(size_t)m * ldO + ncol] = f2bf(v);
      } else if (MODE == 1) {
        ((float*)Out)[(size_t)m * ldO + ncol] = v;
      } else {
        int bb = m / Ldim, l = m % Ldim;
        size_t o = ((size_t)bb * Cdim + ncol) * Ldim + l;
        ((float*)Out)[o] = v + resid[o];
      }
    }
  }
}

// ---------------------------------------------------------------------------
// causal depthwise conv (K=4) + silu on branch-mapped sequence -> bf16 (B*L, Di)
__global__ void conv_kernel(const __bf16* __restrict__ xz, const float* __restrict__ cw,
                            const float* __restrict__ cb, __bf16* __restrict__ xc, int br) {
  int tid = blockIdx.x * blockDim.x + threadIdx.x;
  if (tid >= MROWS * DI) return;
  int e = tid % DI;
  int row = tid / DI;                  // b*L + t (branch order)
  int t = row % LTOT, bb = row / LTOT;
  float acc = cb[e];
#pragma unroll
  for (int j = 0; j < 4; ++j) {
    int tp = t - 3 + j;
    if (tp < 0) continue;
    int src = map_idx(br, tp);
    acc += cw[e * 4 + j] * bf2f(xz[((size_t)bb * LTOT + src) * E2 + e]);
  }
  float s = acc / (1.f + __expf(-acc));            // silu
  xc[(size_t)row * DI + e] = f2bf(s);
}

// dt = softplus(dtr @ dw^T + db): K=16, done in VALU (too small for WMMA K=32)
__global__ void dt_kernel(const float* __restrict__ xdbl, const float* __restrict__ dw,
                          const float* __restrict__ db, float* __restrict__ dt) {
  int tid = blockIdx.x * blockDim.x + threadIdx.x;
  if (tid >= MROWS * DI) return;
  int e = tid % DI, row = tid / DI;
  float acc = db[e];
  const float* xr = xdbl + (size_t)row * XDL;
#pragma unroll
  for (int r = 0; r < DTR; ++r) acc += xr[r] * dw[e * DTR + r];
  dt[(size_t)row * DI + e] = (acc > 20.f) ? acc : log1pf(__expf(acc));
}

// ---------------------------------------------------------------------------
struct ScanArgs {
  const __bf16* xz;
  const __bf16* xc0; const __bf16* xc1; const __bf16* xc2;
  const float*  xd0; const float*  xd1; const float*  xd2;
  const float*  dt0; const float*  dt1; const float*  dt2;
  const float*  al0; const float*  al1; const float*  al2;
  const float*  dp0; const float*  dp1; const float*  dp2;
  float* ysum;
};

// one wave = 2 channels x 16 states; sequential over L; fused gate + un-mapping
__global__ void scan_kernel(ScanArgs s) {
  int w = blockIdx.x * (blockDim.x >> 5) + (threadIdx.x >> 5);
  int lane = threadIdx.x & 31;
  int br  = w / (BB * DI / 2);                 // 512 waves per branch
  int rem = w % (BB * DI / 2);
  int bb  = rem >> 8;                          // 256 waves per batch
  int e   = ((rem & 255) << 1) + (lane >> 4);  // channel
  int n   = lane & 15;                         // state index

  const __bf16* xc = (br == 0) ? s.xc0 : (br == 1) ? s.xc1 : s.xc2;
  const float*  xd = (br == 0) ? s.xd0 : (br == 1) ? s.xd1 : s.xd2;
  const float*  dt = (br == 0) ? s.dt0 : (br == 1) ? s.dt1 : s.dt2;
  const float*  al = (br == 0) ? s.al0 : (br == 1) ? s.al1 : s.al2;
  const float*  dp = (br == 0) ? s.dp0 : (br == 1) ? s.dp1 : s.dp2;

  float A  = -__expf(al[e * NST + n]);
  float Dv = dp[e];
  float h  = 0.f;

  for (int t = 0; t < LTOT; ++t) {
    size_t row = (size_t)bb * LTOT + t;
    float dtv = dt[row * DI + e];
    float u   = bf2f(xc[row * DI + e]);
    float Bv  = xd[row * XDL + DTR + n];
    float Cv  = xd[row * XDL + DTR + NST + n];
    h = h * __expf(dtv * A) + (dtv * u) * Bv;
    float p = h * Cv;
    p += __shfl_xor(p, 1);
    p += __shfl_xor(p, 2);
    p += __shfl_xor(p, 4);
    p += __shfl_xor(p, 8);                      // 16-lane sum over states
    if (n == 0) {
      int oi = map_idx(br, t);                  // un-map to canonical position
      float z = bf2f(s.xz[((size_t)bb * LTOT + oi) * E2 + DI + e]);
      float y = (p + u * Dv) * (z / (1.f + __expf(-z)));
      atomicAdd(&s.ysum[((size_t)bb * LTOT + oi) * DI + e], y);
    }
  }
}

// ---------------------------------------------------------------------------
extern "C" void kernel_launch(void* const* d_in, const int* in_sizes, int n_in,
                              void* d_out, int out_size, void* d_ws, size_t ws_size,
                              hipStream_t stream) {
  (void)in_sizes; (void)n_in; (void)out_size; (void)ws_size;
  const float* x     = (const float*)d_in[0];
  const float* ln_w  = (const float*)d_in[1];
  const float* ln_b  = (const float*)d_in[2];
  const float* inpw  = (const float*)d_in[3];
  const float* outpw = (const float*)d_in[4];
  const float* conv_w[3] = {(const float*)d_in[5],  (const float*)d_in[12], (const float*)d_in[19]};
  const float* conv_b[3] = {(const float*)d_in[6],  (const float*)d_in[13], (const float*)d_in[20]};
  const float* xprojw[3] = {(const float*)d_in[7],  (const float*)d_in[14], (const float*)d_in[21]};
  const float* dtprw[3]  = {(const float*)d_in[8],  (const float*)d_in[15], (const float*)d_in[22]};
  const float* dtprb[3]  = {(const float*)d_in[9],  (const float*)d_in[16], (const float*)d_in[23]};
  const float* alog[3]   = {(const float*)d_in[10], (const float*)d_in[17], (const float*)d_in[24]};
  const float* Dpar[3]   = {(const float*)d_in[11], (const float*)d_in[18], (const float*)d_in[25]};

  // ---- workspace carve (all 256B aligned) --------------------------------
  char* base = (char*)d_ws;
  size_t off = 0;
  auto take = [&](size_t bytes) -> char* {
    char* p = base + off;
    off = (off + bytes + 255) & ~(size_t)255;
    return p;
  };
  __bf16* xn   = (__bf16*)take((size_t)MROWS * CC * 2);
  __bf16* winp = (__bf16*)take((size_t)E2 * CC * 2);
  __bf16* wxp[3]; for (int i = 0; i < 3; ++i) wxp[i] = (__bf16*)take((size_t)XDL * DI * 2);
  __bf16* wout = (__bf16*)take((size_t)CC * DI * 2);
  __bf16* xz   = (__bf16*)take((size_t)MROWS * E2 * 2);
  __bf16* xc[3];   for (int i = 0; i < 3; ++i) xc[i]   = (__bf16*)take((size_t)MROWS * DI * 2);
  float*  xdbl[3]; for (int i = 0; i < 3; ++i) xdbl[i] = (float*)take((size_t)MROWS * XDL * 4);
  float*  dtb[3];  for (int i = 0; i < 3; ++i) dtb[i]  = (float*)take((size_t)MROWS * DI * 4);
  float*  ysum = (float*)take((size_t)MROWS * DI * 4);
  __bf16* ybf  = (__bf16*)take((size_t)MROWS * DI * 2);

  dim3 blk(256);
  int nEl = MROWS * DI;                                // 4,194,304
  zero_f32<<<(nEl + 255) / 256, blk, 0, stream>>>(ysum, nEl);

  // 1) layernorm + transpose -> bf16
  ln_kernel<<<MROWS, CC, 0, stream>>>(x, ln_w, ln_b, xn);

  // 2) weights -> bf16
  cvt_f32_bf16<<<(E2 * CC + 255) / 256, blk, 0, stream>>>(inpw, winp, E2 * CC);
  for (int i = 0; i < 3; ++i)
    cvt_f32_bf16<<<(XDL * DI + 255) / 256, blk, 0, stream>>>(xprojw[i], wxp[i], XDL * DI);
  cvt_f32_bf16<<<(CC * DI + 255) / 256, blk, 0, stream>>>(outpw, wout, CC * DI);

  // 3) in_proj: (8192 x 256) @ (1024 x 256)^T -> xz bf16
  //    blocks = (M/128) * (1024/64) = 64*16 = 1024
  gemm_wmma<0, CC, 4><<<(MROWS / 128) * (E2 / 64), blk, 0, stream>>>(
      xn, winp, xz, CC, E2, E2 / 64, nullptr, 0, 0);

  // 4) per-branch conv+silu, x_proj, dt_proj
  for (int br = 0; br < 3; ++br) {
    conv_kernel<<<(nEl + 255) / 256, blk, 0, stream>>>(xz, conv_w[br], conv_b[br], xc[br], br);
    // x_proj: N=48 = 3 subtiles, one n-tile -> blocks = M/128 = 64
    gemm_wmma<1, DI, 3><<<MROWS / 128, blk, 0, stream>>>(
        xc[br], wxp[br], xdbl[br], DI, XDL, 1, nullptr, 0, 0);
    dt_kernel<<<(nEl + 255) / 256, blk, 0, stream>>>(xdbl[br], dtprw[br], dtprb[br], dtb[br]);
  }

  // 5) selective scan (all 3 branches, fused gate + 3-way sum via atomics)
  ScanArgs sa = { xz,
                  xc[0], xc[1], xc[2],
                  xdbl[0], xdbl[1], xdbl[2],
                  dtb[0], dtb[1], dtb[2],
                  alog[0], alog[1], alog[2],
                  Dpar[0], Dpar[1], Dpar[2],
                  ysum };
  scan_kernel<<<(3 * BB * DI / 2) / 8, blk, 0, stream>>>(sa);   // 1536 waves / 8 per block

  // 6) out_proj + residual -> d_out (B,C,L) fp32
  //    blocks = (M/128) * (256/64) = 64*4 = 256
  cvt_f32_bf16<<<(nEl + 255) / 256, blk, 0, stream>>>(ysum, ybf, nEl);
  gemm_wmma<2, DI, 4><<<(MROWS / 128) * (CC / 64), blk, 0, stream>>>(
      ybf, wout, d_out, DI, 0, CC / 64, x, CC, LTOT);
}
